// LSTM_L_69337952026917
// MI455X (gfx1250) — matrix operand order repaired and checked by reading, not compiled, not run on hardware
//
#include <hip/hip_runtime.h>
#include <hip/hip_bf16.h>
#include <math.h>

typedef __attribute__((ext_vector_type(16))) __bf16 v16bf;
typedef __attribute__((ext_vector_type(8)))  float  v8f;

#define BB 64      // batch
#define SS 512     // sequence length
#define II 1024    // input dim
#define HH 1024    // hidden dim
#define G4 4096    // 4 gates * HH
#define PAD 40     // LDS tile row stride in elements (80 B: bank-conflict-free,
                   // 16-B aligned)

union FragU { uint4 q[2]; v16bf v; };

// ---------------------------------------------------------------------------
// Async staging: copy 16 B (8 bf16) from global to LDS via the CDNA5
// ASYNCcnt-tracked DMA path. dst must be a pointer into __shared__ memory
// (low 32 bits of the generic address are the LDS byte address, ISA §10.2).
// ---------------------------------------------------------------------------
static __device__ __forceinline__ void stage16(const __bf16* __restrict__ base,
                                               int elem_off, __bf16* dst) {
  unsigned lds = (unsigned)(uintptr_t)dst;
  int voff = elem_off * 2;  // byte offset from base
  asm volatile("global_load_async_to_lds_b128 %0, %1, %2"
               :: "v"(lds), "v"(voff), "s"(base) : "memory");
}
// Wait until <= N async ops outstanding (ASYNCcnt completes in order, so this
// retires the older double-buffer chunk while the newer one stays in flight).
template <int N>
static __device__ __forceinline__ void async_wait_le() {
  asm volatile("s_wait_asynccnt %0" :: "i"(N) : "memory");
}

// ---------------------------------------------------------------------------
// WMMA fragment loaders (wave32, V_WMMA_F32_16X16X32_BF16 layouts)
// ---------------------------------------------------------------------------

// A tile in LDS row-major [M][PAD] (M x K). Lanes 0-15: row=lane,
// elems 0..7 = K0..7, 8..15 = K16..23; lanes 16-31: row=lane-16,
// elems 0..7 = K8..15, 8..15 = K24..31. Two contiguous 16-B runs.
static __device__ __forceinline__ v16bf load_a_frag(const __bf16* As,
                                                    int mtile, int lane) {
  int row = mtile * 16 + (lane & 15);
  int kb  = (lane < 16) ? 0 : 8;
  const __bf16* p = As + row * PAD + kb;
  FragU u;
  u.q[0] = *(const uint4*)(p);
  u.q[1] = *(const uint4*)(p + 16);
  return u.v;
}

// B tile in LDS stored K-major per column: [N][PAD]. Lanes 0-15: col=lane,
// K0..7, K16..23; lanes 16-31: col=lane-16, K8..15, K24..31. Two 16-B runs.
static __device__ __forceinline__ v16bf load_b_frag(const __bf16* Bs,
                                                    int ntile, int lane) {
  int col = ntile * 16 + (lane & 15);
  int kb  = (lane < 16) ? 0 : 8;
  const __bf16* p = Bs + col * PAD + kb;
  FragU u;
  u.q[0] = *(const uint4*)(p);
  u.q[1] = *(const uint4*)(p + 16);
  return u.v;
}

// ---------------------------------------------------------------------------
// Generic bf16 GEMM: C(MxN) = A(MxK, row-major) * Bt(NxK, weights stored
// K-major). Block tile 64x128, 256 threads (8 waves); wave w owns N-tile w.
// Double-buffered async LDS staging: issue chunk k+1, wait <=3 (retires
// chunk k), compute chunk k.
// ---------------------------------------------------------------------------
template <bool OUT_BF16>
__global__ __launch_bounds__(256) void gemm_bf16_64x128(
    const __bf16* __restrict__ A, int lda,
    const __bf16* __restrict__ Bt, int ldb,   // ldb = K stride per column
    void* __restrict__ Cout, int ldc, int K) {
  __shared__ __align__(16) __bf16 As[2 * 64 * PAD];
  __shared__ __align__(16) __bf16 Bs[2 * 128 * PAD];
  const int tid  = threadIdx.x;
  const int lane = tid & 31;
  const int wave = tid >> 5;
  const int m0 = blockIdx.y * 64;
  const int n0 = blockIdx.x * 128;
  const int NCH = K >> 5;

  auto issue = [&](int ci) {   // 3 async instrs per wave
    int kk = ci << 5;
    __bf16* Ab = As + (ci & 1) * (64 * PAD);
    __bf16* Bb = Bs + (ci & 1) * (128 * PAD);
    {
      int idx = tid * 8;
      int r = idx >> 5, c = idx & 31;
      stage16(A, (m0 + r) * lda + kk + c, &Ab[r * PAD + c]);
    }
#pragma unroll
    for (int j = 0; j < 2; ++j) {
      int idx = tid * 8 + j * 2048;
      int col = idx >> 5, kc = idx & 31;
      stage16(Bt, (n0 + col) * ldb + kk + kc, &Bb[col * PAD + kc]);
    }
  };

  v8f acc[4] = {};
  issue(0);
  for (int ci = 0; ci < NCH; ++ci) {
    if (ci + 1 < NCH) { issue(ci + 1); async_wait_le<3>(); }
    else              { async_wait_le<0>(); }
    __syncthreads();
    const __bf16* Ab = As + (ci & 1) * (64 * PAD);
    const __bf16* Bb = Bs + (ci & 1) * (128 * PAD);
    v16bf bfrag = load_b_frag(Bb, wave, lane);
#pragma unroll
    for (int mt = 0; mt < 4; ++mt) {
      v16bf afrag = load_a_frag(Ab, mt, lane);
      acc[mt] = __builtin_amdgcn_wmma_f32_16x16x32_bf16(
          false, afrag, false, bfrag, (short)0, acc[mt], false, false);
    }
    __syncthreads();
  }
  const int col = n0 + wave * 16 + (lane & 15);
  const int rbase = (lane < 16) ? 0 : 8;
#pragma unroll
  for (int mt = 0; mt < 4; ++mt) {
#pragma unroll
    for (int r = 0; r < 8; ++r) {
      int row = m0 + mt * 16 + rbase + r;
      if (OUT_BF16)
        ((__bf16*)Cout)[(size_t)row * ldc + col] = (__bf16)acc[mt][r];
      else
        ((float*)Cout)[(size_t)row * ldc + col] = acc[mt][r];
    }
  }
}

// ---------------------------------------------------------------------------
// One LSTM timestep. Grid = HH/64 blocks; block b owns h-columns
// [b*64, b*64+64). Fuses 4 gate GEMMs (K=1024 over h_prev, bf16 WMMA) with
// the x-part add, activations, and the c/h state update for its slice.
// Waves: gate = wave>>1, column half = wave&1 (2 N-tiles x 4 M-tiles each).
// Double-buffered async staging (5 async instrs per wave per chunk).
// ---------------------------------------------------------------------------
__global__ __launch_bounds__(256) void lstm_step(
    const __bf16* __restrict__ hprev,   // 64 x 1024 bf16, row-major
    __bf16* __restrict__ hnext,         // 64 x 1024 bf16
    float* __restrict__ cstate,         // 64 x 1024 f32
    const __bf16* __restrict__ Wht,     // 4096 x 1024 bf16 (K-major weights)
    const __bf16* __restrict__ Xpre,    // 32768 x 4096 bf16, row-major
    int t) {
  // 64 KB LDS: double-buffered A/B staging (51.2 KB) during the GEMM,
  // reused as the 64 KB f32 gate-preact buffer after the final barrier.
  __shared__ __align__(16) unsigned char smem[4 * 64 * 64 * 4];
  __bf16* As2 = (__bf16*)smem;                        // 2 x 64 x PAD
  __bf16* Bs2 = (__bf16*)(smem + 2 * 64 * PAD * 2);   // 2 x 256 x PAD
  float*  Pre = (float*)smem;                         // 4 x 64 x 64 (after)

  const int tid  = threadIdx.x;
  const int lane = tid & 31;
  const int wave = tid >> 5;
  const int g    = wave >> 1;        // gate: 0=i 1=f 2=o 3=g
  const int nh   = wave & 1;         // 32-col half within the 64-col slice
  const int n0   = blockIdx.x * 64;  // h-column tile origin

  auto issue = [&](int ci) {   // 5 async instrs per wave
    int kk = ci << 5;
    __bf16* Ab = As2 + (ci & 1) * (64 * PAD);
    __bf16* Bb = Bs2 + (ci & 1) * (256 * PAD);
    {
      int idx = tid * 8;
      int r = idx >> 5, c = idx & 31;
      stage16(hprev, r * 1024 + kk + c, &Ab[r * PAD + c]);
    }
#pragma unroll
    for (int j = 0; j < 4; ++j) {
      int idx = tid * 8 + j * 2048;
      int gg = idx >> 11;
      int rem = idx & 2047;
      int col = rem >> 5, kc = rem & 31;
      stage16(Wht, (gg * 1024 + n0 + col) * 1024 + kk + kc,
              &Bb[(gg * 64 + col) * PAD + kc]);
    }
  };

  v8f acc[4][2] = {};
  issue(0);
  for (int ci = 0; ci < 32; ++ci) {
    if (ci + 1 < 32) { issue(ci + 1); async_wait_le<5>(); }
    else             { async_wait_le<0>(); }
    __syncthreads();
    const __bf16* Ab = As2 + (ci & 1) * (64 * PAD);
    const __bf16* Bb = Bs2 + (ci & 1) * (256 * PAD);
#pragma unroll
    for (int nt = 0; nt < 2; ++nt) {
      v16bf bfrag = load_b_frag(Bb + g * 64 * PAD, nh * 2 + nt, lane);
#pragma unroll
      for (int mt = 0; mt < 4; ++mt) {
        v16bf afrag = load_a_frag(Ab, mt, lane);
        acc[mt][nt] = __builtin_amdgcn_wmma_f32_16x16x32_bf16(
            false, afrag, false, bfrag, (short)0, acc[mt][nt], false, false);
      }
    }
    __syncthreads();
  }

  // Park gate preacts in LDS (overlaps staging buffers — safe: K-loop ended
  // with a barrier, writes disjoint per wave).
  const int rbase = (lane < 16) ? 0 : 8;
#pragma unroll
  for (int mt = 0; mt < 4; ++mt) {
#pragma unroll
    for (int nt = 0; nt < 2; ++nt) {
      int col = (nh * 2 + nt) * 16 + (lane & 15);
#pragma unroll
      for (int r = 0; r < 8; ++r) {
        int row = mt * 16 + rbase + r;   // batch index
        Pre[(g * 64 + row) * 64 + col] = acc[mt][nt][r];
      }
    }
  }
  __syncthreads();

  // Elementwise: add x-part preacts, activations, c/h update.
  // Each thread owns 16 CONTIGUOUS elements of the 64x64 slice so Pre/Xpre/c/h
  // accesses are vector-width (b128) per lane.
  {
    const int i0 = tid * 16;
    const int row = i0 >> 6;           // batch index (16 elems stay in-row)
    const int c0  = i0 & 63;
    const __bf16* xp = &Xpre[((size_t)row * SS + t) * G4 + n0 + c0];
    float* cptr = &cstate[(size_t)row * 1024 + n0 + c0];
    __bf16* hptr = &hnext[(size_t)row * 1024 + n0 + c0];
#pragma unroll
    for (int e = 0; e < 16; ++e) {
      int col = c0 + e;
      float pi = Pre[(0 * 64 + row) * 64 + col] + (float)xp[0 * 1024 + e];
      float pf = Pre[(1 * 64 + row) * 64 + col] + (float)xp[1 * 1024 + e];
      float po = Pre[(2 * 64 + row) * 64 + col] + (float)xp[2 * 1024 + e];
      float pg = Pre[(3 * 64 + row) * 64 + col] + (float)xp[3 * 1024 + e];
      float iv = 1.0f / (1.0f + __expf(-pi));
      float fv = 1.0f / (1.0f + __expf(-pf));
      float ov = 1.0f / (1.0f + __expf(-po));
      float gv = tanhf(pg);
      float cn = fv * cptr[e] + iv * gv;
      cptr[e] = cn;
      hptr[e] = (__bf16)(ov * tanhf(cn));
    }
  }
}

// ---------------------------------------------------------------------------
// One-time conversion / init kernels
// ---------------------------------------------------------------------------
__global__ void k_f32_to_bf16(const float* __restrict__ s,
                              __bf16* __restrict__ d, int n) {
  int i = blockIdx.x * blockDim.x + threadIdx.x;
  if (i < n) d[i] = (__bf16)s[i];
}

// d[c*rows + r] = s[r*1024 + c] : bf16 transpose (cols fixed at 1024)
__global__ void k_transpose_to_bf16(const float* __restrict__ s,
                                    __bf16* __restrict__ d, int rows) {
  int i = blockIdx.x * blockDim.x + threadIdx.x;
  if (i >= rows * 1024) return;
  int r = i >> 10, c = i & 1023;
  d[(size_t)c * rows + r] = (__bf16)s[i];
}

// Concatenate 4 gate weight matrices (rows x 1024 each, row-major f32) into
// one transposed bf16 matrix: d[(g*1024+n)*rows + k] = s_g[k*1024 + n].
__global__ void k_concat4_t(const float* __restrict__ s0,
                            const float* __restrict__ s1,
                            const float* __restrict__ s2,
                            const float* __restrict__ s3,
                            __bf16* __restrict__ d, int rows) {
  int i = blockIdx.x * blockDim.x + threadIdx.x;   // i = k*4096 + (g*1024+n)
  if (i >= rows * G4) return;
  int k = i >> 12, cc = i & (G4 - 1);
  int g = cc >> 10, n = cc & 1023;
  const float* s = (g == 0) ? s0 : (g == 1) ? s1 : (g == 2) ? s2 : s3;
  d[(size_t)cc * rows + k] = (__bf16)s[(size_t)k * 1024 + n];
}

__global__ void k_zero_f32(float* __restrict__ p, int n) {
  int i = blockIdx.x * blockDim.x + threadIdx.x;
  if (i < n) p[i] = 0.0f;
}

// ---------------------------------------------------------------------------
// Launch
// ---------------------------------------------------------------------------
extern "C" void kernel_launch(void* const* d_in, const int* in_sizes, int n_in,
                              void* d_out, int out_size, void* d_ws,
                              size_t ws_size, hipStream_t stream) {
  (void)in_sizes; (void)n_in; (void)out_size; (void)ws_size;
  const float* x    = (const float*)d_in[0];
  const float* W_hi = (const float*)d_in[1];
  const float* W_ii = (const float*)d_in[2];
  const float* W_hf = (const float*)d_in[3];
  const float* W_if = (const float*)d_in[4];
  const float* W_ho = (const float*)d_in[5];
  const float* W_io = (const float*)d_in[6];
  const float* W_hg = (const float*)d_in[7];
  const float* W_ig = (const float*)d_in[8];
  const float* W_y  = (const float*)d_in[9];

  char* ws = (char*)d_ws;
  size_t off = 0;
  auto wsalloc = [&](size_t bytes) -> char* {
    char* p = ws + off;
    off += (bytes + 255) & ~(size_t)255;
    return p;
  };
  __bf16* xbf  = (__bf16*)wsalloc((size_t)BB * SS * II * 2);  //  64 MB
  __bf16* Wit  = (__bf16*)wsalloc((size_t)II * G4 * 2);       //   8 MB (4096x1024)
  __bf16* Wht  = (__bf16*)wsalloc((size_t)HH * G4 * 2);       //   8 MB (4096x1024)
  __bf16* Wyt  = (__bf16*)wsalloc((size_t)HH * HH * 2);       //   2 MB (1024x1024)
  __bf16* Xpre = (__bf16*)wsalloc((size_t)BB * SS * G4 * 2);  // 256 MB
  __bf16* h0   = (__bf16*)wsalloc((size_t)BB * HH * 2);
  __bf16* h1   = (__bf16*)wsalloc((size_t)BB * HH * 2);
  float*  cst  = (float*)wsalloc((size_t)BB * HH * 4);

  const int nx = BB * SS * II;
  k_f32_to_bf16<<<(nx + 255) / 256, 256, 0, stream>>>(x, xbf, nx);
  k_concat4_t<<<(II * G4 + 255) / 256, 256, 0, stream>>>(W_ii, W_if, W_io,
                                                         W_ig, Wit, II);
  k_concat4_t<<<(HH * G4 + 255) / 256, 256, 0, stream>>>(W_hi, W_hf, W_ho,
                                                         W_hg, Wht, HH);
  k_transpose_to_bf16<<<(HH * HH + 255) / 256, 256, 0, stream>>>(W_y, Wyt, HH);
  k_zero_f32<<<(BB * HH / 2 + 255) / 256, 256, 0, stream>>>((float*)h0,
                                                            BB * HH / 2);
  k_zero_f32<<<(BB * HH + 255) / 256, 256, 0, stream>>>(cst, BB * HH);

  // Hoisted input projections: Xpre = X(32768x1024) @ Wi(1024x4096), bf16 out.
  gemm_bf16_64x128<true>
      <<<dim3(G4 / 128, (BB * SS) / 64), 256, 0, stream>>>(
          xbf, II, Wit, II, (void*)Xpre, G4, II);

  // Sequential recurrence: 512 fused gate-GEMM + state-update steps.
  __bf16* hb[2] = {h0, h1};
  for (int t = 0; t < SS; ++t) {
    lstm_step<<<HH / 64, 256, 0, stream>>>(hb[t & 1], hb[(t + 1) & 1], cst,
                                           Wht, Xpre, t);
  }

  // Output projection: y = h_T(64x1024) @ Wy(1024x1024), f32 out to d_out.
  gemm_bf16_64x128<false>
      <<<dim3(HH / 128, BB / 64), 256, 0, stream>>>(
          hb[SS & 1], HH, Wyt, HH, d_out, HH, HH);
}
